// GCNLSTM_23321672417514
// MI455X (gfx1250) — matrix operand order repaired and checked by reading, not compiled, not run on hardware
//
#include <hip/hip_runtime.h>

#define NN 16000
#define EE 256000
#define TT 5
#define HD 64
#define LHD 128
#define GG 512        // 4*LH gate width
#define GGP (GG + 4)  // padded row to avoid LDS bank conflicts

typedef __attribute__((ext_vector_type(16))) __bf16 v16bf;
typedef __attribute__((ext_vector_type(8)))  float  v8f;

// fast sigmoid / tanh: v_exp_f32 + v_rcp_f32 (no IEEE-division expansion)
__device__ __forceinline__ float sigf(float x) {
  return __builtin_amdgcn_rcpf(1.0f + __expf(-x));
}
__device__ __forceinline__ float tanh_fast(float x) {
  return fmaf(2.0f, sigf(2.0f * x), -1.0f);
}

// Map element (m, j) of a 16xK bf16 A-matrix into the CDNA5 16-bit A-fragment
// layout (16x32 tiles): VGPR0 lanes0-15 K=0,1; lanes16-31 K=8,9; VGPR4-7 K=16..31.
// Flattened as [kt][lane][slot], 16 bf16 per lane per tile.
__device__ __forceinline__ int afrag_idx(int m, int j) {
  int kt = j >> 5, kk = j & 31;
  int grp = kk >> 4, r = kk & 15, hf = r >> 3, q = r & 7;
  int lane = hf * 16 + m;
  int slot = ((grp * 4 + (q >> 1)) << 1) + (q & 1);
  return ((kt * 32 + lane) << 4) + slot;
}

__device__ __forceinline__ v16bf ldfrag(const __bf16* p) { return *(const v16bf*)p; }

__device__ __forceinline__ v8f wmma_bf16(v16bf a, v16bf b, v8f c) {
  return __builtin_amdgcn_wmma_f32_16x16x32_bf16(false, a, false, b, (short)0, c, false, false);
}

// ---------------- GCN scalar-aggregation prep ----------------

__global__ void deg_init_kernel(float* deg) {
  int i = blockIdx.x * blockDim.x + threadIdx.x;
  if (i < NN) deg[i] = 1.0f;  // self-loop
}

__global__ void deg_count_kernel(const int* __restrict__ ei, float* deg) {
  int e = blockIdx.x * blockDim.x + threadIdx.x;
  if (e < EE) atomicAdd(&deg[ei[EE + e]], 1.0f);
}

__global__ void dinv_kernel(const float* __restrict__ deg, float* dinv) {
  int i = blockIdx.x * blockDim.x + threadIdx.x;
  if (i < NN) dinv[i] = rsqrtf(deg[i]);  // deg >= 1 always
}

__global__ void agg_init_kernel(const float* __restrict__ x, const float* __restrict__ dinv,
                                float* agg) {
  int i = blockIdx.x * blockDim.x + threadIdx.x;
  if (i >= TT * NN) return;
  int t = i / NN, n = i - t * NN;
  agg[i] = dinv[n] * dinv[n] * x[n * TT + t];  // self-loop term (overwrites poison)
}

__global__ void agg_edge_kernel(const float* __restrict__ x, const int* __restrict__ ei,
                                const float* __restrict__ dinv, float* agg) {
  int e = blockIdx.x * blockDim.x + threadIdx.x;
  if (e >= EE) return;
  int s = ei[e], d = ei[EE + e];
  float nrm = dinv[s] * dinv[d];
#pragma unroll
  for (int t = 0; t < TT; ++t) atomicAdd(&agg[t * NN + d], nrm * x[s * TT + t]);
}

__global__ void bias_combine_kernel(const float* __restrict__ a, const float* __restrict__ b,
                                    float* o) {
  int i = blockIdx.x * blockDim.x + threadIdx.x;
  if (i < GG) o[i] = a[i] + b[i];
}

// Pack W (GG x K, f32 row-major) into WMMA B-fragment tiles (32x16, 512 bf16 each),
// interleaved per layer: tile index = nt*KtTot + ktOff + kt, so each column tile's
// whole K reduction is one contiguous burst (L2-resident stream).
__global__ void pack_w_kernel(const float* __restrict__ W, __bf16* out, int K, int ktOff,
                              int KtTot) {
  int i = blockIdx.x * blockDim.x + threadIdx.x;
  if (i >= GG * K) return;
  int nout = i / K, k = i - nout * K;
  int nt = nout >> 4, n = nout & 15;
  int kt = k >> 5, kk = k & 31;
  int lane = (((kk >> 3) & 1) << 4) + n;
  int slot = ((kk >> 4) << 3) + (kk & 7);
  out[(((size_t)(nt * KtTot + ktOff + kt) * 32 + lane) << 4) + slot] = (__bf16)W[i];
}

// ---------------- fused 2-layer LSTM + head ----------------

__global__ void __launch_bounds__(256) fused_lstm_kernel(
    const float* __restrict__ x, const float* __restrict__ agg,
    const float* __restrict__ gcn_w, const float* __restrict__ gcn_b,
    const __bf16* __restrict__ pwL0,  // layer0: 32 nt x 6 ktiles x 512 bf16
    const __bf16* __restrict__ pwL1,  // layer1: 32 nt x 8 ktiles x 512 bf16
    const float* __restrict__ bias0, const float* __restrict__ bias1,
    const float* __restrict__ pw1, const float* __restrict__ pb1,
    const float* __restrict__ pw2, const float* __restrict__ pb2,
    float* __restrict__ out) {
  __shared__ __bf16 aX[2 * 32 * 16];    // layer0 input frags (16x64)
  __shared__ __bf16 aH0[4 * 32 * 16];   // h0 frags (16x128) == layer1 input frags
  __shared__ __bf16 aH1[4 * 32 * 16];   // h1 frags (16x128)
  __shared__ float gbuf[16][GGP];       // gate pre-activations (padded rows)
  __shared__ float c0[16][LHD];
  __shared__ float c1[16][LHD];
  __shared__ float zbuf[16][64];
  __shared__ float gwS[HD], gbS[HD];

  const int tid = threadIdx.x;
  const int lane = tid & 31;
  const int wv = tid >> 5;  // 8 waves
  const int node0 = blockIdx.x * 16;

  for (int i = tid; i < 4 * 32 * 16; i += 256) { aH0[i] = (__bf16)0.0f; aH1[i] = (__bf16)0.0f; }
  for (int i = tid; i < 16 * LHD; i += 256) { c0[i >> 7][i & 127] = 0.0f; c1[i >> 7][i & 127] = 0.0f; }
  for (int i = tid; i < HD; i += 256) { gwS[i] = gcn_w[i]; gbS[i] = gcn_b[i]; }
  __syncthreads();

#pragma clang loop unroll(disable)
  for (int t = 0; t < TT; ++t) {
    // GCN activation -> layer0 A fragments (bf16)
    for (int idx = tid; idx < 16 * HD; idx += 256) {
      int m = idx >> 6, k = idx & 63;
      float a = agg[t * NN + node0 + m];
      float v = fmaxf(fmaf(a, gwS[k], gbS[k]), 0.0f);
      aX[afrag_idx(m, k)] = (__bf16)v;
    }
    __syncthreads();

    // ---- layer 0 gates: (16x64)@Wih0^T + (16x128)@Whh0^T + bias ----
    {
      // A fragments in named registers (kept live across the nt loop)
      const v16bf ax0 = ldfrag(aX + ((0 * 32 + lane) << 4));
      const v16bf ax1 = ldfrag(aX + ((1 * 32 + lane) << 4));
      const v16bf ah0 = ldfrag(aH0 + ((0 * 32 + lane) << 4));
      const v16bf ah1 = ldfrag(aH0 + ((1 * 32 + lane) << 4));
      const v16bf ah2 = ldfrag(aH0 + ((2 * 32 + lane) << 4));
      const v16bf ah3 = ldfrag(aH0 + ((3 * 32 + lane) << 4));
#pragma clang loop unroll(disable)
      for (int nti = 0; nti < 4; ++nti) {
        int nt = (wv << 2) + nti;
        int nc = (nt << 4) + (lane & 15);
        float bv = bias0[nc];
        v8f acc;
#pragma unroll
        for (int r = 0; r < 8; ++r) acc[r] = bv;
        const __bf16* bp = pwL0 + (((size_t)nt * 6 * 32 + lane) << 4);
        acc = wmma_bf16(ax0, ldfrag(bp + 0 * 512), acc);
        acc = wmma_bf16(ax1, ldfrag(bp + 1 * 512), acc);
        acc = wmma_bf16(ah0, ldfrag(bp + 2 * 512), acc);
        acc = wmma_bf16(ah1, ldfrag(bp + 3 * 512), acc);
        acc = wmma_bf16(ah2, ldfrag(bp + 4 * 512), acc);
        acc = wmma_bf16(ah3, ldfrag(bp + 5 * 512), acc);
        int mrow = (lane >> 4) << 3;
#pragma unroll
        for (int r = 0; r < 8; ++r) gbuf[mrow + r][nc] = acc[r];
      }
    }
    __syncthreads();

    // layer0 cell; write h0 straight into A-fragment layout (doubles as layer1 input)
    for (int idx = tid; idx < 16 * LHD; idx += 256) {
      int m = idx >> 7, j = idx & 127;
      float gi = gbuf[m][j], gf = gbuf[m][LHD + j];
      float gg = gbuf[m][2 * LHD + j], go = gbuf[m][3 * LHD + j];
      float cc = sigf(gf) * c0[m][j] + sigf(gi) * tanh_fast(gg);
      float hh = sigf(go) * tanh_fast(cc);
      c0[m][j] = cc;
      aH0[afrag_idx(m, j)] = (__bf16)hh;
    }
    __syncthreads();

    // ---- layer 1 gates: input = h0 frags, recurrent = h1 frags ----
    {
      const v16bf ai0 = ldfrag(aH0 + ((0 * 32 + lane) << 4));
      const v16bf ai1 = ldfrag(aH0 + ((1 * 32 + lane) << 4));
      const v16bf ai2 = ldfrag(aH0 + ((2 * 32 + lane) << 4));
      const v16bf ai3 = ldfrag(aH0 + ((3 * 32 + lane) << 4));
      const v16bf ar0 = ldfrag(aH1 + ((0 * 32 + lane) << 4));
      const v16bf ar1 = ldfrag(aH1 + ((1 * 32 + lane) << 4));
      const v16bf ar2 = ldfrag(aH1 + ((2 * 32 + lane) << 4));
      const v16bf ar3 = ldfrag(aH1 + ((3 * 32 + lane) << 4));
#pragma clang loop unroll(disable)
      for (int nti = 0; nti < 4; ++nti) {
        int nt = (wv << 2) + nti;
        int nc = (nt << 4) + (lane & 15);
        float bv = bias1[nc];
        v8f acc;
#pragma unroll
        for (int r = 0; r < 8; ++r) acc[r] = bv;
        const __bf16* bp = pwL1 + (((size_t)nt * 8 * 32 + lane) << 4);
        acc = wmma_bf16(ai0, ldfrag(bp + 0 * 512), acc);
        acc = wmma_bf16(ai1, ldfrag(bp + 1 * 512), acc);
        acc = wmma_bf16(ai2, ldfrag(bp + 2 * 512), acc);
        acc = wmma_bf16(ai3, ldfrag(bp + 3 * 512), acc);
        acc = wmma_bf16(ar0, ldfrag(bp + 4 * 512), acc);
        acc = wmma_bf16(ar1, ldfrag(bp + 5 * 512), acc);
        acc = wmma_bf16(ar2, ldfrag(bp + 6 * 512), acc);
        acc = wmma_bf16(ar3, ldfrag(bp + 7 * 512), acc);
        int mrow = (lane >> 4) << 3;
#pragma unroll
        for (int r = 0; r < 8; ++r) gbuf[mrow + r][nc] = acc[r];
      }
    }
    __syncthreads();

    // layer1 cell
    for (int idx = tid; idx < 16 * LHD; idx += 256) {
      int m = idx >> 7, j = idx & 127;
      float gi = gbuf[m][j], gf = gbuf[m][LHD + j];
      float gg = gbuf[m][2 * LHD + j], go = gbuf[m][3 * LHD + j];
      float cc = sigf(gf) * c1[m][j] + sigf(gi) * tanh_fast(gg);
      float hh = sigf(go) * tanh_fast(cc);
      c1[m][j] = cc;
      aH1[afrag_idx(m, j)] = (__bf16)hh;
    }
    __syncthreads();
  }

  // head: relu(h1 @ pw1^T + pb1) @ pw2^T + pb2 ; out = x[:, -1] + delta
  for (int idx = tid; idx < 16 * 64; idx += 256) {
    int m = idx >> 6, u = idx & 63;
    float s = pb1[u];
    for (int j = 0; j < LHD; ++j)
      s = fmaf((float)aH1[afrag_idx(m, j)], pw1[u * LHD + j], s);
    zbuf[m][u] = fmaxf(s, 0.0f);
  }
  __syncthreads();
  if (tid < 16) {
    int m = tid;
    float s = pb2[0];
    for (int u = 0; u < 64; ++u) s = fmaf(zbuf[m][u], pw2[u], s);
    out[node0 + m] = x[(node0 + m) * TT + (TT - 1)] + s;
  }
}

// ---------------- launch ----------------

extern "C" void kernel_launch(void* const* d_in, const int* in_sizes, int n_in,
                              void* d_out, int out_size, void* d_ws, size_t ws_size,
                              hipStream_t stream) {
  (void)in_sizes; (void)n_in; (void)out_size; (void)ws_size;
  const float* x     = (const float*)d_in[0];
  const int*   ei    = (const int*)d_in[1];
  const float* gcn_w = (const float*)d_in[2];
  const float* gcn_b = (const float*)d_in[3];
  const float* w_ih0 = (const float*)d_in[4];
  const float* w_hh0 = (const float*)d_in[5];
  const float* b_ih0 = (const float*)d_in[6];
  const float* b_hh0 = (const float*)d_in[7];
  const float* w_ih1 = (const float*)d_in[8];
  const float* w_hh1 = (const float*)d_in[9];
  const float* b_ih1 = (const float*)d_in[10];
  const float* b_hh1 = (const float*)d_in[11];
  const float* pw1   = (const float*)d_in[12];
  const float* pb1   = (const float*)d_in[13];
  const float* pw2   = (const float*)d_in[14];
  const float* pb2   = (const float*)d_in[15];
  float* out = (float*)d_out;

  size_t off = 0;
  auto alloc = [&](size_t bytes) -> void* {
    off = (off + 255) & ~(size_t)255;
    void* p = (char*)d_ws + off;
    off += bytes;
    return p;
  };
  float*  deg   = (float*)alloc((size_t)NN * 4);
  float*  dinv  = (float*)alloc((size_t)NN * 4);
  float*  agg   = (float*)alloc((size_t)TT * NN * 4);
  __bf16* pwL0  = (__bf16*)alloc((size_t)GG * (HD + LHD) * 2);   // 6 ktiles/nt
  __bf16* pwL1  = (__bf16*)alloc((size_t)GG * (LHD + LHD) * 2);  // 8 ktiles/nt
  float*  bias0 = (float*)alloc((size_t)GG * 4);
  float*  bias1 = (float*)alloc((size_t)GG * 4);

  deg_init_kernel<<<(NN + 255) / 256, 256, 0, stream>>>(deg);
  deg_count_kernel<<<(EE + 255) / 256, 256, 0, stream>>>(ei, deg);
  dinv_kernel<<<(NN + 255) / 256, 256, 0, stream>>>(deg, dinv);
  agg_init_kernel<<<(TT * NN + 255) / 256, 256, 0, stream>>>(x, dinv, agg);
  agg_edge_kernel<<<(EE + 255) / 256, 256, 0, stream>>>(x, ei, dinv, agg);
  bias_combine_kernel<<<(GG + 255) / 256, 256, 0, stream>>>(b_ih0, b_hh0, bias0);
  bias_combine_kernel<<<(GG + 255) / 256, 256, 0, stream>>>(b_ih1, b_hh1, bias1);
  pack_w_kernel<<<(GG * HD + 255) / 256, 256, 0, stream>>>(w_ih0, pwL0, HD, 0, 6);
  pack_w_kernel<<<(GG * LHD + 255) / 256, 256, 0, stream>>>(w_hh0, pwL0, LHD, 2, 6);
  pack_w_kernel<<<(GG * LHD + 255) / 256, 256, 0, stream>>>(w_ih1, pwL1, LHD, 0, 8);
  pack_w_kernel<<<(GG * LHD + 255) / 256, 256, 0, stream>>>(w_hh1, pwL1, LHD, 4, 8);

  fused_lstm_kernel<<<NN / 16, 256, 0, stream>>>(
      x, agg, gcn_w, gcn_b, pwL0, pwL1, bias0, bias1, pw1, pb1, pw2, pb2, out);
}
